// SelectiveSSM_87282325389775
// MI455X (gfx1250) — matrix operand order repaired and checked by reading, not compiled, not run on hardware
//
#include <hip/hip_runtime.h>
#include <hip/hip_bf16.h>
#include <stdint.h>

// ---------------------------------------------------------------------------
// Mamba-style selective SSM block for MI455X (gfx1250), wave32 + WMMA.
// GEMMs: bf16 inputs, f32 accumulate via v_wmma_f32_16x16x32_bf16,
//        double-buffered LDS fed by global_load_async_to_lds_b128 (ASYNCcnt),
//        manually unrolled ping-pong so buffer indices are compile-time.
// Scan: 16 state lanes per channel, shfl_xor reduction, fused gating.
// ---------------------------------------------------------------------------

typedef __attribute__((ext_vector_type(16))) __bf16    v16bf;
typedef __attribute__((ext_vector_type(8)))  float     v8f;
typedef __attribute__((ext_vector_type(4)))  unsigned  u32x4;

union FragU { v16bf v; u32x4 q[2]; };

#define USE_ASYNC_LDS 1

__device__ __forceinline__ void copy16_async(const unsigned short* g, unsigned short* l) {
#if USE_ASYNC_LDS
    // LDS aperture keeps the wave-relative LDS offset in addr[31:0].
    unsigned lds_off = (unsigned)(uintptr_t)l;
    asm volatile("global_load_async_to_lds_b128 %0, %1, off"
                 :: "v"(lds_off), "v"(g)
                 : "memory");
#else
    *(u32x4*)l = *(const u32x4*)g;
#endif
}

__device__ __forceinline__ void wait_async_copies() {
#if USE_ASYNC_LDS
#if defined(__has_builtin)
#if __has_builtin(__builtin_amdgcn_s_wait_asynccnt)
    __builtin_amdgcn_s_wait_asynccnt(0);
#else
    asm volatile("s_wait_asynccnt 0x0" ::: "memory");
#endif
#else
    asm volatile("s_wait_asynccnt 0x0" ::: "memory");
#endif
#endif
}

__device__ __forceinline__ unsigned short f2bf(float f) {
    unsigned u = __float_as_uint(f);
    unsigned r = (u + 0x7FFFu + ((u >> 16) & 1u)) >> 16;
    return (unsigned short)r;
}

// ---- fp32 -> bf16 convert (row-major, no transpose) -----------------------
__global__ void k_cvt_bf16(const float* __restrict__ in,
                           unsigned short* __restrict__ out, int n) {
    int i = blockIdx.x * blockDim.x + threadIdx.x;
    if (i < n) out[i] = f2bf(in[i]);
}

// ---- fp32 (K x N) -> bf16 transposed (N x K) ------------------------------
__global__ void k_cvt_bf16_t(const float* __restrict__ in,
                             unsigned short* __restrict__ out, int K, int N) {
    int i = blockIdx.x * blockDim.x + threadIdx.x;
    if (i >= K * N) return;
    int k = i / N, n = i % N;
    out[(size_t)n * K + k] = f2bf(in[i]);
}

// ---- WMMA GEMM: C(M,N) = A(M,K) * B(K,N) + bias, fused epilogues ---------
// A: bf16 row-major (M x K).  Bt: bf16, weights pre-transposed (N x K).
// Workgroup tile 128x128, BK=32, 8 waves (2x4), per-wave 64x32 = 8 WMMA accums.
// mode 0: split columns at splitN -> out0 (xc_raw), out1 (res)
// mode 1: softplus(v + bias) -> out0   (delta)
// mode 2: v + bias -> out0             (final output)
__global__ __launch_bounds__(256)
void k_gemm_wmma(const unsigned short* __restrict__ Abf,
                 const unsigned short* __restrict__ Btbf,
                 const float* __restrict__ bias,
                 float* __restrict__ out0, float* __restrict__ out1,
                 int M, int Nn, int Kd, int mode, int splitN) {
    // rows padded to 40 ushorts (80B) for bank-conflict-free fragment reads
    __shared__ unsigned short Al[2][128 * 40];
    __shared__ unsigned short Bl[2][128 * 40];

    const int tid   = threadIdx.x;
    const int wave  = tid >> 5;
    const int lane  = tid & 31;
    const int waveM = wave >> 2;      // 0..1  (64 rows each)
    const int waveN = wave & 3;       // 0..3  (32 cols each)
    const int blockN0 = blockIdx.x * 128;
    const int blockM0 = blockIdx.y * 128;

    v8f acc[4][2] = {};

    const int lh  = lane & 15;
    const int klo = (lane < 16) ? 0 : 8;    // A-frag K low-group base
    const int kb  = (lane < 16) ? 0 : 16;   // unused; B frags also read at klo

    // staging: A tile 128x32 and B tile 128x32, 512 16B chunks each -> 2/thread
    const int sRow0 = tid >> 2;             // 0..63
    const int sRow1 = sRow0 + 64;           // 64..127
    const int sCk   = (tid & 3) * 8;        // 0,8,16,24
    const unsigned short* Ag0 = Abf  + (size_t)(blockM0 + sRow0) * Kd + sCk;
    const unsigned short* Ag1 = Abf  + (size_t)(blockM0 + sRow1) * Kd + sCk;
    const unsigned short* Bg0 = Btbf + (size_t)(blockN0 + sRow0) * Kd + sCk;
    const unsigned short* Bg1 = Btbf + (size_t)(blockN0 + sRow1) * Kd + sCk;

    auto stage = [&](int buf, int kk) {
        copy16_async(Ag0 + kk, &Al[buf][sRow0 * 40 + sCk]);
        copy16_async(Ag1 + kk, &Al[buf][sRow1 * 40 + sCk]);
        copy16_async(Bg0 + kk, &Bl[buf][sRow0 * 40 + sCk]);
        copy16_async(Bg1 + kk, &Bl[buf][sRow1 * 40 + sCk]);
    };

    auto compute = [&](int buf) {
        FragU a[4], b[2];
#pragma unroll
        for (int mt = 0; mt < 4; ++mt) {
            int mrow = waveM * 64 + mt * 16 + lh;
            a[mt].q[0] = *(const u32x4*)(&Al[buf][mrow * 40 + klo]);        // K {0-7 | 8-15}
            a[mt].q[1] = *(const u32x4*)(&Al[buf][mrow * 40 + 16 + klo]);   // K {16-23 | 24-31}
        }
#pragma unroll
        for (int nt = 0; nt < 2; ++nt) {
            int nrow = waveN * 32 + nt * 16 + lh;
            b[nt].q[0] = *(const u32x4*)(&Bl[buf][nrow * 40 + klo]);        // K {0-7 | 8-15}
            b[nt].q[1] = *(const u32x4*)(&Bl[buf][nrow * 40 + 16 + klo]);   // K {16-23 | 24-31}
        }
#pragma unroll
        for (int mt = 0; mt < 4; ++mt)
#pragma unroll
            for (int nt = 0; nt < 2; ++nt)
                acc[mt][nt] = __builtin_amdgcn_wmma_f32_16x16x32_bf16(
                    false, a[mt].v, false, b[nt].v, (short)0, acc[mt][nt],
                    false, false);
    };

    // ---- software pipeline: Kd is a multiple of 64 (Kd = 1024 here) ----
    stage(0, 0);
    int k0 = 0;
    for (; k0 < Kd - 64; k0 += 64) {
        wait_async_copies();          // buf0 slab landed
        __syncthreads();
        stage(1, k0 + 32);
        compute(0);
        wait_async_copies();          // buf1 slab landed
        __syncthreads();
        stage(0, k0 + 64);
        compute(1);
    }
    // peeled tail: slabs k0 (buf0) and k0+32 (buf1)
    wait_async_copies();
    __syncthreads();
    stage(1, k0 + 32);
    compute(0);
    wait_async_copies();
    __syncthreads();
    compute(1);

    // epilogue: C layout lane<16 -> M row r, lane>=16 -> M row 8+r; N = lane&15
    const int mhi = (lane >> 4) * 8;
#pragma unroll
    for (int mt = 0; mt < 4; ++mt)
        for (int nt = 0; nt < 2; ++nt) {
            int nG = blockN0 + waveN * 32 + nt * 16 + lh;
            float bv = bias[nG];
#pragma unroll
            for (int r = 0; r < 8; ++r) {
                int mG = blockM0 + waveM * 64 + mt * 16 + mhi + r;
                float v = acc[mt][nt][r] + bv;
                if (mode == 0) {
                    if (nG < splitN) out0[(size_t)mG * splitN + nG] = v;
                    else             out1[(size_t)mG * splitN + (nG - splitN)] = v;
                } else if (mode == 1) {
                    float sp = (v > 20.f) ? v : __logf(1.f + __expf(v));
                    out0[(size_t)mG * Nn + nG] = sp;
                } else {
                    out0[(size_t)mG * Nn + nG] = v;
                }
            }
        }
}

// ---- depthwise causal conv (K=4) + SiLU -----------------------------------
__global__ void k_conv_silu(const float* __restrict__ xcraw,
                            const float* __restrict__ cw,
                            const float* __restrict__ cb,
                            float* __restrict__ xc, int Bb, int L, int D) {
    int idx = blockIdx.x * blockDim.x + threadIdx.x;
    if (idx >= Bb * L * D) return;
    int d = idx % D, ml = idx / D, l = ml % L, b = ml / L;
    float acc = cb[d];
#pragma unroll
    for (int k = 0; k < 4; ++k) {
        int lk = l - 3 + k;
        if (lk >= 0) acc += cw[d * 4 + k] * xcraw[((size_t)b * L + lk) * D + d];
    }
    xc[idx] = acc / (1.f + __expf(-acc));   // silu
}

// ---- x_proj: (4096 x 1024) @ (1024 x 32) -> B,C ---------------------------
__global__ void k_xproj(const float* __restrict__ xc,
                        const float* __restrict__ w,
                        const float* __restrict__ bias,
                        float* __restrict__ bc, int Kd) {
    int m = blockIdx.x, t = threadIdx.x;
    int j = t & 31, part = t >> 5;                // 8 K-partitions
    const float* xr = xc + (size_t)m * Kd;
    float acc = 0.f;
    int k0 = part * 128;
    for (int k = k0; k < k0 + 128; ++k) acc += xr[k] * w[k * 32 + j];
    __shared__ float red[256];
    red[t] = acc;
    __syncthreads();
    if (t < 32) {
        float s = bias[j];
#pragma unroll
        for (int p = 0; p < 8; ++p) s += red[j + p * 32];
        bc[(size_t)m * 32 + j] = s;
    }
}

// ---- selective scan: 16 state-lanes per (b,d) channel, fused gating -------
__global__ void k_scan(const float* __restrict__ delta,
                       const float* __restrict__ xc,
                       const float* __restrict__ bc,
                       const float* __restrict__ res,
                       const float* __restrict__ A_log,
                       const float* __restrict__ Dp,
                       unsigned short* __restrict__ zbf, int L, int D) {
    int g = blockIdx.x * blockDim.x + threadIdx.x;
    int n  = g & 15;
    int bd = g >> 4;
    int d  = bd & (D - 1);
    int b  = bd / D;
    float An = -__expf(A_log[n]);
    float Dd = Dp[d];
    float h = 0.f;
    const size_t base = (size_t)b * L;
#pragma unroll 2
    for (int t = 0; t < L; ++t) {
        size_t m = base + t;
        float dt = delta[m * D + d];
        float u  = xc[m * D + d];
        float Bn = bc[m * 32 + n];
        float Cn = bc[m * 32 + 16 + n];
        float dA = __expf(dt * An);
        h = dA * h + (dt * u) * Bn;
        float s = h * Cn;
        s += __shfl_xor(s, 1);
        s += __shfl_xor(s, 2);
        s += __shfl_xor(s, 4);
        s += __shfl_xor(s, 8);
        if (n == 0) {
            float y = s + u * Dd;
            float r = res[m * D + d];
            float gate = r / (1.f + __expf(-r));   // silu(res)
            zbf[m * D + d] = f2bf(y * gate);
        }
    }
}

// ---------------------------------------------------------------------------
extern "C" void kernel_launch(void* const* d_in, const int* in_sizes, int n_in,
                              void* d_out, int out_size, void* d_ws, size_t ws_size,
                              hipStream_t stream) {
    const int Bb = 2, L = 2048, D = 1024;
    const int M = Bb * L;                 // 4096
    const size_t MB = 1u << 20;

    const float* x         = (const float*)d_in[0];
    const float* in_proj_w = (const float*)d_in[1];
    const float* in_proj_b = (const float*)d_in[2];
    const float* conv_w    = (const float*)d_in[3];
    const float* conv_b    = (const float*)d_in[4];
    const float* x_proj_w  = (const float*)d_in[5];
    const float* x_proj_b  = (const float*)d_in[6];
    const float* dt_proj_w = (const float*)d_in[7];
    const float* dt_proj_b = (const float*)d_in[8];
    const float* A_log     = (const float*)d_in[9];
    const float* D_param   = (const float*)d_in[10];
    const float* out_proj_w= (const float*)d_in[11];
    const float* out_proj_b= (const float*)d_in[12];

    char* ws = (char*)d_ws;
    float*          xcraw = (float*)(ws + 0 * MB);    // 16 MB; reused for delta
    float*          res   = (float*)(ws + 16 * MB);   // 16 MB
    float*          xc    = (float*)(ws + 32 * MB);   // 16 MB
    unsigned short* abf   = (unsigned short*)(ws + 48 * MB); // 8 MB: x_bf / xc_bf / z_bf
    unsigned short* wbf   = (unsigned short*)(ws + 56 * MB); // 4 MB: weight bf16 (transposed)
    float*          bc    = (float*)(ws + 60 * MB);   // 0.5 MB
    float*          delta = xcraw;

    const int MD = M * D;                 // 4,194,304

    // 1) in_proj GEMM (M=4096, N=2048, K=1024) -> xc_raw | res
    k_cvt_bf16_t<<<(D * 2 * D + 255) / 256, 256, 0, stream>>>(in_proj_w, wbf, D, 2 * D);
    k_cvt_bf16  <<<MD / 256, 256, 0, stream>>>(x, abf, MD);
    k_gemm_wmma <<<dim3(16, 32), 256, 0, stream>>>(abf, wbf, in_proj_b,
                                                   xcraw, res, M, 2 * D, D, 0, D);
    // 2) depthwise conv + silu
    k_conv_silu <<<MD / 256, 256, 0, stream>>>(xcraw, conv_w, conv_b, xc, Bb, L, D);
    // 3) x_proj -> B, C
    k_cvt_bf16  <<<MD / 256, 256, 0, stream>>>(xc, abf, MD);
    k_xproj     <<<M, 256, 0, stream>>>(xc, x_proj_w, x_proj_b, bc, D);
    // 4) dt_proj GEMM + softplus -> delta
    k_cvt_bf16_t<<<(D * D + 255) / 256, 256, 0, stream>>>(dt_proj_w, wbf, D, D);
    k_gemm_wmma <<<dim3(8, 32), 256, 0, stream>>>(abf, wbf, dt_proj_b,
                                                  delta, nullptr, M, D, D, 1, D);
    // 5) selective scan + gating -> z (bf16, reuses abf)
    k_scan      <<<(Bb * D * 16) / 256, 256, 0, stream>>>(delta, xc, bc, res,
                                                          A_log, D_param, abf, L, D);
    // 6) out_proj GEMM -> d_out
    k_cvt_bf16_t<<<(D * D + 255) / 256, 256, 0, stream>>>(out_proj_w, wbf, D, D);
    k_gemm_wmma <<<dim3(8, 32), 256, 0, stream>>>(abf, wbf, out_proj_b,
                                                  (float*)d_out, nullptr, M, D, D, 2, D);
}
